// NucleusMoEImageTransformerBlock_69733089017995
// MI455X (gfx1250) — matrix operand order repaired and checked by reading, not compiled, not run on hardware
//
#include <hip/hip_runtime.h>
#include <hip/hip_bf16.h>
#include <stdint.h>

#define B_   2
#define SI_  1024
#define ST_  512
#define D_   2048
#define H_   16
#define KV_  4
#define HD_  128
#define SKV_ 1536
#define E_   8
#define F_   1024
#define T_   2048   // B*SI tokens

#define LDS_STRIDE 40   // 32 + 8 pad halves; 80B row, 16B aligned

typedef __attribute__((ext_vector_type(16))) __bf16 v16bf;
typedef __attribute__((ext_vector_type(8)))  float  v8f;

union V16U { v16bf v; uint4 q[2]; unsigned short s[16]; };

__device__ __forceinline__ unsigned short f2bu(float f) {
  union { float f; uint32_t u; } c; c.f = f;
  uint32_t u = c.u;
  return (unsigned short)((u + 0x7fffu + ((u >> 16) & 1u)) >> 16);  // RNE f32->bf16
}

// ---- CDNA5 async global->LDS copy (ASYNCcnt-tracked, no VGPR round-trip) ----
__device__ __forceinline__ void async_b128(uint32_t ldsOff, const void* g) {
  asm volatile("global_load_async_to_lds_b128 %0, %1, off"
               :: "v"(ldsOff), "v"(g) : "memory");
}
#define WAIT_ASYNC(n) asm volatile("s_wait_asynccnt " #n ::: "memory")

__device__ __forceinline__ uint32_t lds_off(const void* p) {
  return (uint32_t)(uintptr_t)p;   // low 32 bits of generic shared addr = LDS offset
}

// 16x32 bf16 fragment from a contiguous 32-half row slice (A-layout)
__device__ __forceinline__ v16bf frag_row(const unsigned short* rowPtr, int hi) {
  V16U u;
  u.q[0] = *(const uint4*)(rowPtr + hi * 8);
  u.q[1] = *(const uint4*)(rowPtr + 16 + hi * 8);
  return u.v;
}
// B-layout fragment (32x16) from a [n][k] row slice
__device__ __forceinline__ v16bf frag_bt(const unsigned short* rowPtr, int hi) {
  V16U u;
  u.q[0] = *(const uint4*)(rowPtr + hi * 8);
  u.q[1] = *(const uint4*)(rowPtr + 16 + hi * 8);
  return u.v;
}

__device__ __forceinline__ v8f wmma_bf16(v16bf a, v16bf b, v8f c) {
  return __builtin_amdgcn_wmma_f32_16x16x32_bf16(false, a, false, b, (short)0, c, false, false);
}

// ---------------- conversion / norm kernels ----------------

__global__ void __launch_bounds__(256)
k_f32_to_bf16(const float* __restrict__ in, unsigned short* __restrict__ out, long long n) {
  long long i = (long long)blockIdx.x * 256 + threadIdx.x;
  if (i < n) out[i] = f2bu(in[i]);
}

// batched tiled transpose+convert: in[b][R][Cc] f32 -> out[b][Cc][R] bf16
__global__ void __launch_bounds__(256)
k_transpose_bf16(const float* __restrict__ in, unsigned short* __restrict__ out, int R, int Cc) {
  __shared__ unsigned short tile[32][33];
  const size_t bsz = (size_t)R * Cc;
  const float* ib = in + (size_t)blockIdx.z * bsz;
  unsigned short* ob = out + (size_t)blockIdx.z * bsz;
  const int c0 = blockIdx.x * 32, r0 = blockIdx.y * 32;
  const int tx = threadIdx.x & 31, ty = threadIdx.x >> 5;
#pragma unroll
  for (int i = 0; i < 32; i += 8)
    tile[ty + i][tx] = f2bu(ib[(size_t)(r0 + ty + i) * Cc + (c0 + tx)]);
  __syncthreads();
#pragma unroll
  for (int i = 0; i < 32; i += 8)
    ob[(size_t)(c0 + ty + i) * R + (r0 + tx)] = tile[tx][ty + i];
}

__global__ void __launch_bounds__(256)
k_rmsnorm_bf16(const float* __restrict__ x, const float* __restrict__ w,
               unsigned short* __restrict__ out, int D) {
  const int row = blockIdx.x;
  const float* xr = x + (size_t)row * D;
  float ss = 0.f;
  for (int i = threadIdx.x; i < D; i += 256) { float v = xr[i]; ss += v * v; }
#pragma unroll
  for (int d = 16; d >= 1; d >>= 1) ss += __shfl_xor(ss, d, 32);
  __shared__ float red[8];
  __shared__ float rs_s;
  if ((threadIdx.x & 31) == 0) red[threadIdx.x >> 5] = ss;
  __syncthreads();
  if (threadIdx.x == 0) {
    float tot = 0.f;
    for (int i = 0; i < 8; ++i) tot += red[i];
    rs_s = rsqrtf(tot / (float)D + 1e-6f);
  }
  __syncthreads();
  const float rs = rs_s;
  for (int i = threadIdx.x; i < D; i += 256)
    out[(size_t)row * D + i] = f2bu(xr[i] * rs * w[i]);
}

// per-(b,s,head) RMS over HD=128 + RoPE, bf16 strided output
__global__ void __launch_bounds__(64)
k_rms_rope(const float* __restrict__ x, const float* __restrict__ nw,
           const float* __restrict__ cosT, const float* __restrict__ sinT,
           unsigned short* __restrict__ out,
           int heads, int S, long long outBStride, int outRowStride, int seqOff) {
  const int id = blockIdx.x;
  const int head = id % heads;
  const int s = (id / heads) % S;
  const int b = id / (heads * S);
  const float* xr = x + (size_t)id * HD_;
  const int t = threadIdx.x;  // pair index 0..63
  float a0 = xr[2 * t], a1 = xr[2 * t + 1];
  float ss = a0 * a0 + a1 * a1;
#pragma unroll
  for (int d = 16; d >= 1; d >>= 1) ss += __shfl_xor(ss, d, 32);
  __shared__ float red2[2];
  if ((t & 31) == 0) red2[t >> 5] = ss;
  __syncthreads();
  const float rs = rsqrtf((red2[0] + red2[1]) / (float)HD_ + 1e-6f);
  float y0 = a0 * rs * nw[2 * t];
  float y1 = a1 * rs * nw[2 * t + 1];
  float c = cosT[(size_t)s * (HD_ / 2) + t], sn = sinT[(size_t)s * (HD_ / 2) + t];
  unsigned short* dst = out + (size_t)b * outBStride +
                        (size_t)(seqOff + s) * outRowStride + (size_t)head * HD_;
  dst[2 * t]     = f2bu(y0 * c - y1 * sn);
  dst[2 * t + 1] = f2bu(y0 * sn + y1 * c);
}

__global__ void __launch_bounds__(256)
k_copy_v(const float* __restrict__ in, unsigned short* __restrict__ out,
         int S, long long outBStride, int seqOff) {
  const int row = blockIdx.x;  // b*S + s
  const int b = row / S, s = row % S;
  const float* src = in + (size_t)row * (KV_ * HD_);
  unsigned short* dst = out + (size_t)b * outBStride + (size_t)(seqOff + s) * (KV_ * HD_);
  for (int i = threadIdx.x; i < KV_ * HD_; i += 256) dst[i] = f2bu(src[i]);
}

// ---------------- tiled bf16 WMMA GEMM (async double-buffered) ----------------
// C[M,N](f32) = A[M,K](bf16,row-major) @ Bt[N,K](bf16,row-major = B^T) (+residual)
__global__ void __launch_bounds__(256)
k_gemm(const unsigned short* __restrict__ A, const unsigned short* __restrict__ Bt,
       const float* __restrict__ residual, float* __restrict__ C,
       int M, int N, int K) {
  __shared__ __align__(16) unsigned short As[2][128 * LDS_STRIDE];
  __shared__ __align__(16) unsigned short Bs[2][128 * LDS_STRIDE];
  const int wave = threadIdx.x >> 5, lane = threadIdx.x & 31;
  const int hi = lane >> 4, cl = lane & 15;
  const int wm = wave >> 2, wn = wave & 3;
  const int m_base = blockIdx.y * 128;
  const int n_base = blockIdx.x * 128;

  v8f acc[4][2];
  if (residual) {
#pragma unroll
    for (int i = 0; i < 4; ++i)
#pragma unroll
      for (int j = 0; j < 2; ++j) {
        const int rowb = m_base + wm * 64 + i * 16 + 8 * hi;
        const int colb = n_base + wn * 32 + j * 16 + cl;
#pragma unroll
        for (int r = 0; r < 8; ++r)
          acc[i][j][r] = residual[(size_t)(rowb + r) * N + colb];
      }
  } else {
#pragma unroll
    for (int i = 0; i < 4; ++i)
#pragma unroll
      for (int j = 0; j < 2; ++j)
#pragma unroll
        for (int r = 0; r < 8; ++r) acc[i][j][r] = 0.f;
  }

  const int sr = threadIdx.x >> 1;            // 0..127 staging row
  const int sc = (threadIdx.x & 1) * 16;      // 0 or 16
  const unsigned short* aSrc = A  + (size_t)(m_base + sr) * K + sc;
  const unsigned short* bSrc = Bt + (size_t)(n_base + sr) * K + sc;
  uint32_t aOff[2], bOff[2];
#pragma unroll
  for (int u = 0; u < 2; ++u) {
    aOff[u] = lds_off(&As[u][sr * LDS_STRIDE + sc]);
    bOff[u] = lds_off(&Bs[u][sr * LDS_STRIDE + sc]);
  }
  auto stage = [&](int k0, int buf) {
    async_b128(aOff[buf],      aSrc + k0);
    async_b128(aOff[buf] + 16, aSrc + k0 + 8);
    async_b128(bOff[buf],      bSrc + k0);
    async_b128(bOff[buf] + 16, bSrc + k0 + 8);
  };

  const int nsteps = K / 32;
  stage(0, 0);
  for (int s = 0; s < nsteps; ++s) {
    const int buf = s & 1;
    if (s + 1 < nsteps) {
      stage((s + 1) * 32, buf ^ 1);
      __builtin_prefetch(aSrc + (s + 2) * 32, 0, 1);   // global_prefetch_b8
      __builtin_prefetch(bSrc + (s + 2) * 32, 0, 1);
      WAIT_ASYNC(0x4);          // older tile's 4 copies complete (in-order)
    } else {
      WAIT_ASYNC(0x0);
    }
    __syncthreads();

    v16bf af[4], bfr[2];
#pragma unroll
    for (int i = 0; i < 4; ++i)
      af[i] = frag_row(&As[buf][(wm * 64 + i * 16 + cl) * LDS_STRIDE], hi);
#pragma unroll
    for (int j = 0; j < 2; ++j)
      bfr[j] = frag_bt(&Bs[buf][(wn * 32 + j * 16 + cl) * LDS_STRIDE], hi);
#pragma unroll
    for (int i = 0; i < 4; ++i)
#pragma unroll
      for (int j = 0; j < 2; ++j)
        acc[i][j] = wmma_bf16(af[i], bfr[j], acc[i][j]);
    __syncthreads();
  }

#pragma unroll
  for (int i = 0; i < 4; ++i)
#pragma unroll
    for (int j = 0; j < 2; ++j) {
      const int rowb = m_base + wm * 64 + i * 16 + 8 * hi;
      const int colb = n_base + wn * 32 + j * 16 + cl;
#pragma unroll
      for (int r = 0; r < 8; ++r)
        C[(size_t)(rowb + r) * N + colb] = acc[i][j][r];
    }
}

// ---------------- flash attention (WMMA, async double-buffered K/V) ----------------
__global__ void __launch_bounds__(128)
k_attn(const unsigned short* __restrict__ qb, const unsigned short* __restrict__ kb,
       const unsigned short* __restrict__ vb, unsigned short* __restrict__ ob) {
  __shared__ __align__(16) unsigned short Kt[2][32 * HD_];
  __shared__ __align__(16) unsigned short Vt[2][32 * HD_];
  __shared__ __align__(16) float Ps[4][16 * 32];

  const int blk = blockIdx.x;
  const int qt = blk & 15;
  const int h = (blk >> 4) & 15;
  const int b = blk >> 8;
  const int wave = threadIdx.x >> 5, lane = threadIdx.x & 31;
  const int hi = lane >> 4, cl = lane & 15;
  const int kvh = h >> 2;
  const int q0 = qt * 64 + wave * 16;

  const unsigned short* qRow = qb + (size_t)((b * SI_ + q0 + cl) * H_ + h) * HD_;
  v16bf qf[4];
#pragma unroll
  for (int kk = 0; kk < 4; ++kk) qf[kk] = frag_row(qRow + kk * 32, hi);

  float m[8], l[8];
  v8f O[8];
#pragma unroll
  for (int r = 0; r < 8; ++r) { m[r] = -1e30f; l[r] = 0.f; }
#pragma unroll
  for (int nt = 0; nt < 8; ++nt)
#pragma unroll
    for (int r = 0; r < 8; ++r) O[nt][r] = 0.f;

  const float scale = 0.088388347648318447f;  // 1/sqrt(128)

  auto stageKV = [&](int s0, int buf) {
#pragma unroll
    for (int j = 0; j < 4; ++j) {
      int chunk = threadIdx.x + 128 * j;
      int row = chunk >> 4, cc = chunk & 15;
      size_t g = (size_t)((b * SKV_ + s0 + row) * KV_ + kvh) * HD_ + cc * 8;
      async_b128(lds_off(&Kt[buf][row * HD_ + cc * 8]), kb + g);
      async_b128(lds_off(&Vt[buf][row * HD_ + cc * 8]), vb + g);
    }
  };

  stageKV(0, 0);
  for (int s0 = 0, step = 0; s0 < SKV_; s0 += 32, ++step) {
    const int buf = step & 1;
    if (s0 + 32 < SKV_) {
      stageKV(s0 + 32, buf ^ 1);
      WAIT_ASYNC(0x8);          // older tile's 8 copies complete
    } else {
      WAIT_ASYNC(0x0);
    }
    __syncthreads();

    v8f sc0, sc1;
#pragma unroll
    for (int r = 0; r < 8; ++r) { sc0[r] = 0.f; sc1[r] = 0.f; }
#pragma unroll
    for (int kk = 0; kk < 4; ++kk) {   // S = Q @ K^T
      sc0 = wmma_bf16(qf[kk], frag_bt(&Kt[buf][(0 * 16 + cl) * HD_ + kk * 32], hi), sc0);
      sc1 = wmma_bf16(qf[kk], frag_bt(&Kt[buf][(1 * 16 + cl) * HD_ + kk * 32], hi), sc1);
    }

    float fr[8];
#pragma unroll
    for (int r = 0; r < 8; ++r) {
      float a = sc0[r] * scale, bb = sc1[r] * scale;
      float mx = fmaxf(a, bb);
#pragma unroll
      for (int d = 1; d <= 8; d <<= 1) mx = fmaxf(mx, __shfl_xor(mx, d, 32));
      float mn = fmaxf(m[r], mx);
      fr[r] = __expf(m[r] - mn);
      float p0 = __expf(a - mn), p1 = __expf(bb - mn);
      float sum = p0 + p1;
#pragma unroll
      for (int d = 1; d <= 8; d <<= 1) sum += __shfl_xor(sum, d, 32);
      l[r] = l[r] * fr[r] + sum;
      m[r] = mn;
      Ps[wave][(r + 8 * hi) * 32 + cl]      = p0;
      Ps[wave][(r + 8 * hi) * 32 + 16 + cl] = p1;
    }
#pragma unroll
    for (int nt = 0; nt < 8; ++nt)
#pragma unroll
      for (int r = 0; r < 8; ++r) O[nt][r] *= fr[r];

    // rebuild P as A-fragment (vectorized float4 LDS reads, same-wave ordering)
    V16U pu;
    {
      const float* pr = &Ps[wave][cl * 32];
      float4 pA = *(const float4*)(pr + hi * 8);
      float4 pB = *(const float4*)(pr + hi * 8 + 4);
      float4 pC = *(const float4*)(pr + 16 + hi * 8);
      float4 pD = *(const float4*)(pr + 16 + hi * 8 + 4);
      pu.s[0] = f2bu(pA.x);  pu.s[1] = f2bu(pA.y);  pu.s[2] = f2bu(pA.z);  pu.s[3] = f2bu(pA.w);
      pu.s[4] = f2bu(pB.x);  pu.s[5] = f2bu(pB.y);  pu.s[6] = f2bu(pB.z);  pu.s[7] = f2bu(pB.w);
      pu.s[8] = f2bu(pC.x);  pu.s[9] = f2bu(pC.y);  pu.s[10] = f2bu(pC.z); pu.s[11] = f2bu(pC.w);
      pu.s[12] = f2bu(pD.x); pu.s[13] = f2bu(pD.y); pu.s[14] = f2bu(pD.z); pu.s[15] = f2bu(pD.w);
    }

#pragma unroll
    for (int nt = 0; nt < 8; ++nt) {   // O += P @ V
      const unsigned short* vp = &Vt[buf][nt * 16 + cl];
      V16U vu;
#pragma unroll
      for (int i = 0; i < 16; ++i) vu.s[i] = vp[(size_t)((i & 7) + ((i >> 3) << 4) + hi * 8) * HD_];
      O[nt] = wmma_bf16(pu.v, vu.v, O[nt]);
    }
    __syncthreads();
  }

#pragma unroll
  for (int r = 0; r < 8; ++r) {
    float inv = 1.f / l[r];
    int qrow = q0 + r + 8 * hi;
    unsigned short* dst = ob + (size_t)(b * SI_ + qrow) * (H_ * HD_) + h * HD_;
#pragma unroll
    for (int nt = 0; nt < 8; ++nt) dst[nt * 16 + cl] = f2bu(O[nt][r] * inv);
  }
}

// ---------------- router + MoE ----------------

__global__ void k_init(int* counts, int* cursor) {
  if (threadIdx.x < E_) { counts[threadIdx.x] = 0; cursor[threadIdx.x] = 0; }
}

__global__ void __launch_bounds__(256)
k_router(const float* __restrict__ h, const float* __restrict__ nw,
         const float* __restrict__ rw, unsigned short* __restrict__ xt,
         int* __restrict__ topi, float* __restrict__ topw, int* __restrict__ counts) {
  const int t = blockIdx.x;
  const float* hr = h + (size_t)t * D_;
  float ss = 0.f;
  for (int i = threadIdx.x; i < D_; i += 256) { float v = hr[i]; ss += v * v; }
#pragma unroll
  for (int d = 16; d >= 1; d >>= 1) ss += __shfl_xor(ss, d, 32);
  __shared__ float red[8];
  __shared__ float rs_s;
  if ((threadIdx.x & 31) == 0) red[threadIdx.x >> 5] = ss;
  __syncthreads();
  if (threadIdx.x == 0) {
    float tot = 0.f;
    for (int i = 0; i < 8; ++i) tot += red[i];
    rs_s = rsqrtf(tot / (float)D_ + 1e-6f);
  }
  __syncthreads();
  const float rs = rs_s;
  float acc[E_];
#pragma unroll
  for (int e = 0; e < E_; ++e) acc[e] = 0.f;
  for (int i = threadIdx.x; i < D_; i += 256) {
    float v = hr[i] * rs * nw[i];
    xt[(size_t)t * D_ + i] = f2bu(v);
#pragma unroll
    for (int e = 0; e < E_; ++e) acc[e] += v * rw[(size_t)i * E_ + e];
  }
#pragma unroll
  for (int e = 0; e < E_; ++e)
#pragma unroll
    for (int d = 16; d >= 1; d >>= 1) acc[e] += __shfl_xor(acc[e], d, 32);
  __shared__ float lred[8][E_];
  if ((threadIdx.x & 31) == 0)
    for (int e = 0; e < E_; ++e) lred[threadIdx.x >> 5][e] = acc[e];
  __syncthreads();
  if (threadIdx.x == 0) {
    float lg[E_];
    for (int e = 0; e < E_; ++e) { float s = 0.f; for (int w = 0; w < 8; ++w) s += lred[w][e]; lg[e] = s; }
    float mx = lg[0];
    for (int e = 1; e < E_; ++e) mx = fmaxf(mx, lg[e]);
    float p[E_], sum = 0.f;
    for (int e = 0; e < E_; ++e) { p[e] = __expf(lg[e] - mx); sum += p[e]; }
    for (int e = 0; e < E_; ++e) p[e] /= sum;
    int i0 = 0;
    for (int e = 1; e < E_; ++e) if (p[e] > p[i0]) i0 = e;
    int i1 = (i0 == 0) ? 1 : 0;
    for (int e = 0; e < E_; ++e) if (e != i0 && p[e] > p[i1]) i1 = e;
    float inv = 1.f / (p[i0] + p[i1]);
    topi[t * 2] = i0; topi[t * 2 + 1] = i1;
    topw[t * 2] = p[i0] * inv; topw[t * 2 + 1] = p[i1] * inv;
    atomicAdd(&counts[i0], 1); atomicAdd(&counts[i1], 1);
  }
}

__global__ void k_offsets(const int* counts, int* offs) {
  if (threadIdx.x == 0) {
    int s = 0;
    for (int e = 0; e < E_; ++e) { offs[e] = s; s += counts[e]; }
    offs[E_] = s;
  }
}

__global__ void __launch_bounds__(256)
k_scatter(const int* __restrict__ topi, const int* __restrict__ offs,
          int* cursor, int* __restrict__ perm) {
  int t = blockIdx.x * 256 + threadIdx.x;
  if (t >= T_) return;
#pragma unroll
  for (int slot = 0; slot < 2; ++slot) {
    int e = topi[t * 2 + slot];
    int pos = atomicAdd(&cursor[e], 1);
    perm[offs[e] + pos] = t * 2 + slot;   // encodes token*2 + slot
  }
}

// gate_up: gathered rows of xt @ gupT[e] (Bt layout [2F][D])
__global__ void __launch_bounds__(256)
k_moe_gate_up(const unsigned short* __restrict__ xt, const unsigned short* __restrict__ gupT,
              const int* __restrict__ counts, const int* __restrict__ offs,
              const int* __restrict__ perm, float* __restrict__ gu) {
  __shared__ __align__(16) unsigned short As[2][128 * LDS_STRIDE];
  __shared__ __align__(16) unsigned short Bs[2][128 * LDS_STRIDE];
  const int e = blockIdx.z;
  const int cnt = counts[e];
  const int m0 = blockIdx.y * 128;
  if (m0 >= cnt) return;
  const int off = offs[e];
  const int wave = threadIdx.x >> 5, lane = threadIdx.x & 31;
  const int hi = lane >> 4, cl = lane & 15;
  const int wm = wave >> 2, wn = wave & 3;
  const int n_base = blockIdx.x * 128;

  const int sr = threadIdx.x >> 1;
  const int sc = (threadIdx.x & 1) * 16;
  int mr = m0 + sr; if (mr >= cnt) mr = cnt - 1;
  const int tok = perm[off + mr] >> 1;
  const unsigned short* aSrc = xt + (size_t)tok * D_ + sc;
  const unsigned short* bSrc = gupT + (size_t)e * (2 * F_) * D_ + (size_t)(n_base + sr) * D_ + sc;
  uint32_t aOff[2], bOff[2];
#pragma unroll
  for (int u = 0; u < 2; ++u) {
    aOff[u] = lds_off(&As[u][sr * LDS_STRIDE + sc]);
    bOff[u] = lds_off(&Bs[u][sr * LDS_STRIDE + sc]);
  }
  auto stage = [&](int k0, int buf) {
    async_b128(aOff[buf],      aSrc + k0);
    async_b128(aOff[buf] + 16, aSrc + k0 + 8);
    async_b128(bOff[buf],      bSrc + k0);
    async_b128(bOff[buf] + 16, bSrc + k0 + 8);
  };

  v8f acc[4][2];
#pragma unroll
  for (int i = 0; i < 4; ++i)
#pragma unroll
    for (int j = 0; j < 2; ++j)
#pragma unroll
      for (int r = 0; r < 8; ++r) acc[i][j][r] = 0.f;

  const int nsteps = D_ / 32;
  stage(0, 0);
  for (int s = 0; s < nsteps; ++s) {
    const int buf = s & 1;
    if (s + 1 < nsteps) {
      stage((s + 1) * 32, buf ^ 1);
      WAIT_ASYNC(0x4);
    } else {
      WAIT_ASYNC(0x0);
    }
    __syncthreads();
    v16bf af[4], bfr[2];
#pragma unroll
    for (int i = 0; i < 4; ++i)
      af[i] = frag_row(&As[buf][(wm * 64 + i * 16 + cl) * LDS_STRIDE], hi);
#pragma unroll
    for (int j = 0; j < 2; ++j)
      bfr[j] = frag_bt(&Bs[buf][(wn * 32 + j * 16 + cl) * LDS_STRIDE], hi);
#pragma unroll
    for (int i = 0; i < 4; ++i)
#pragma unroll
      for (int j = 0; j < 2; ++j)
        acc[i][j] = wmma_bf16(af[i], bfr[j], acc[i][j]);
    __syncthreads();
  }

#pragma unroll
  for (int i = 0; i < 4; ++i)
#pragma unroll
    for (int j = 0; j < 2; ++j) {
      const int rowb = m0 + wm * 64 + i * 16 + 8 * hi;
      const int colb = n_base + wn * 32 + j * 16 + cl;
#pragma unroll
      for (int r = 0; r < 8; ++r)
        if (rowb + r < cnt) gu[(size_t)(off + rowb + r) * (2 * F_) + colb] = acc[i][j][r];
    }
}

__global__ void __launch_bounds__(256)
k_silu(const float* __restrict__ gu, unsigned short* __restrict__ act) {
  int i = blockIdx.x * 256 + threadIdx.x;
  if (i >= T_ * 2 * F_ / 2) return;          // 2048 slots * F
  int slot = i >> 10, f = i & (F_ - 1);
  float g = gu[(size_t)slot * (2 * F_) + f];
  float u = gu[(size_t)slot * (2 * F_) + F_ + f];
  act[i] = f2bu((g / (1.f + __expf(-g))) * u);
}

// down: act rows @ dwnT[e] (Bt layout [D][F]); scatter rows via perm
__global__ void __launch_bounds__(256)
k_moe_down(const unsigned short* __restrict__ act, const unsigned short* __restrict__ dwnT,
           const int* __restrict__ counts, const int* __restrict__ offs,
           const int* __restrict__ perm, float* __restrict__ eo) {
  __shared__ __align__(16) unsigned short As[2][128 * LDS_STRIDE];
  __shared__ __align__(16) unsigned short Bs[2][128 * LDS_STRIDE];
  const int e = blockIdx.z;
  const int cnt = counts[e];
  const int m0 = blockIdx.y * 128;
  if (m0 >= cnt) return;
  const int off = offs[e];
  const int wave = threadIdx.x >> 5, lane = threadIdx.x & 31;
  const int hi = lane >> 4, cl = lane & 15;
  const int wm = wave >> 2, wn = wave & 3;
  const int n_base = blockIdx.x * 128;

  const int sr = threadIdx.x >> 1;
  const int sc = (threadIdx.x & 1) * 16;
  int mr = m0 + sr; if (mr >= cnt) mr = cnt - 1;
  const unsigned short* aSrc = act + (size_t)(off + mr) * F_ + sc;
  const unsigned short* bSrc = dwnT + (size_t)e * D_ * F_ + (size_t)(n_base + sr) * F_ + sc;
  uint32_t aOff[2], bOff[2];
#pragma unroll
  for (int u = 0; u < 2; ++u) {
    aOff[u] = lds_off(&As[u][sr * LDS_STRIDE + sc]);
    bOff[u] = lds_off(&Bs[u][sr * LDS_STRIDE + sc]);
  }
  auto stage = [&](int k0, int buf) {
    async_b128(aOff[buf],      aSrc + k0);
    async_b128(aOff[buf] + 16, aSrc + k0 + 8);
    async_b128(bOff[buf],      bSrc + k0);
    async_b128(bOff[buf] + 16, bSrc + k0 + 8);
  };

  v8f acc[4][2];
#pragma unroll
  for (int i = 0; i < 4; ++i)
#pragma unroll
    for (int j = 0; j < 2; ++j)
#pragma unroll
      for (int r = 0; r < 8; ++r) acc[i][j][r] = 0.f;

  const int nsteps = F_ / 32;
  stage(0, 0);
  for (int s = 0; s < nsteps; ++s) {
    const int buf = s & 1;
    if (s + 1 < nsteps) {
      stage((s + 1) * 32, buf ^ 1);
      WAIT_ASYNC(0x4);
    } else {
      WAIT_ASYNC(0x0);
    }
    __syncthreads();
    v16bf af[4], bfr[2];
#pragma unroll
    for (int i = 0; i < 4; ++i)
      af[i] = frag_row(&As[buf][(wm * 64 + i * 16 + cl) * LDS_STRIDE], hi);
#pragma unroll
    for (int j = 0; j < 2; ++j)
      bfr[j] = frag_bt(&Bs[buf][(wn * 32 + j * 16 + cl) * LDS_STRIDE], hi);
#pragma unroll
    for (int i = 0; i < 4; ++i)
#pragma unroll
      for (int j = 0; j < 2; ++j)
        acc[i][j] = wmma_bf16(af[i], bfr[j], acc[i][j]);
    __syncthreads();
  }

#pragma unroll
  for (int i = 0; i < 4; ++i)
#pragma unroll
    for (int j = 0; j < 2; ++j) {
      const int rowb = m0 + wm * 64 + i * 16 + 8 * hi;
      const int colb = n_base + wn * 32 + j * 16 + cl;
#pragma unroll
      for (int r = 0; r < 8; ++r)
        if (rowb + r < cnt) {
          int slot = perm[off + rowb + r];   // token*2 + slot index
          eo[(size_t)slot * D_ + colb] = acc[i][j][r];
        }
    }
}

__global__ void __launch_bounds__(256)
k_final(float* __restrict__ out, const float* __restrict__ eo, const float* __restrict__ topw) {
  int i = blockIdx.x * 256 + threadIdx.x;
  if (i >= T_ * D_) return;
  int t = i >> 11, d = i & (D_ - 1);
  out[i] += topw[t * 2]     * eo[(size_t)(t * 2) * D_ + d]
          + topw[t * 2 + 1] * eo[(size_t)(t * 2 + 1) * D_ + d];
}

// ---------------- host side ----------------

extern "C" void kernel_launch(void* const* d_in, const int* in_sizes, int n_in,
                              void* d_out, int out_size, void* d_ws, size_t ws_size,
                              hipStream_t stream) {
  (void)in_sizes; (void)n_in; (void)out_size; (void)ws_size;
  const float* hidden   = (const float*)d_in[0];
  const float* enc      = (const float*)d_in[1];
  const float* img_cos  = (const float*)d_in[2];
  const float* img_sin  = (const float*)d_in[3];
  const float* txt_cos  = (const float*)d_in[4];
  const float* txt_sin  = (const float*)d_in[5];
  const float* norm1_w  = (const float*)d_in[6];
  const float* norm2_w  = (const float*)d_in[7];
  const float* wq       = (const float*)d_in[8];
  const float* wk       = (const float*)d_in[9];
  const float* wv       = (const float*)d_in[10];
  const float* wk_txt   = (const float*)d_in[11];
  const float* wv_txt   = (const float*)d_in[12];
  const float* wo       = (const float*)d_in[13];
  const float* q_norm_w = (const float*)d_in[14];
  const float* k_norm_w = (const float*)d_in[15];
  const float* akn_w    = (const float*)d_in[16];
  const float* router_w = (const float*)d_in[17];
  const float* gup      = (const float*)d_in[18];
  const float* dwn      = (const float*)d_in[19];
  float* out = (float*)d_out;

  char* p = (char*)d_ws;
  auto alloc = [&](size_t bytes) -> char* {
    char* r = p; p += (bytes + 255) & ~(size_t)255; return r;
  };
  typedef unsigned short u16;
  u16*   x_bf   = (u16*)alloc((size_t)T_ * D_ * 2);
  u16*   enc_bf = (u16*)alloc((size_t)B_ * ST_ * D_ * 2);
  u16*   wqT    = (u16*)alloc((size_t)D_ * (H_ * HD_) * 2);   // [2048][D]
  u16*   wkT    = (u16*)alloc((size_t)D_ * (KV_ * HD_) * 2);  // [512][D]
  u16*   wvT    = (u16*)alloc((size_t)D_ * (KV_ * HD_) * 2);
  u16*   wktT   = (u16*)alloc((size_t)D_ * (KV_ * HD_) * 2);
  u16*   wvtT   = (u16*)alloc((size_t)D_ * (KV_ * HD_) * 2);
  u16*   woT    = (u16*)alloc((size_t)(H_ * HD_) * D_ * 2);   // [D][2048]
  u16*   gupT   = (u16*)alloc((size_t)E_ * D_ * (2 * F_) * 2); // [E][2F][D]
  u16*   dwnT   = (u16*)alloc((size_t)E_ * F_ * D_ * 2);       // [E][D][F]
  float* qf     = (float*)alloc((size_t)T_ * (H_ * HD_) * 4);
  float* kimgf  = (float*)alloc((size_t)T_ * (KV_ * HD_) * 4);
  float* vimgf  = (float*)alloc((size_t)T_ * (KV_ * HD_) * 4);
  float* ktxtf  = (float*)alloc((size_t)B_ * ST_ * (KV_ * HD_) * 4);
  float* vtxtf  = (float*)alloc((size_t)B_ * ST_ * (KV_ * HD_) * 4);
  u16*   q_bf   = (u16*)alloc((size_t)T_ * (H_ * HD_) * 2);
  u16*   k_bf   = (u16*)alloc((size_t)B_ * SKV_ * KV_ * HD_ * 2);
  u16*   v_bf   = (u16*)alloc((size_t)B_ * SKV_ * KV_ * HD_ * 2);
  u16*   o_bf   = (u16*)alloc((size_t)T_ * (H_ * HD_) * 2);
  u16*   xt_bf  = (u16*)alloc((size_t)T_ * D_ * 2);
  int*   topi   = (int*)alloc((size_t)T_ * 2 * 4);
  float* topw   = (float*)alloc((size_t)T_ * 2 * 4);
  int*   counts = (int*)alloc(64);
  int*   cursor = (int*)alloc(64);
  int*   offs   = (int*)alloc(64);
  int*   perm   = (int*)alloc((size_t)T_ * 2 * 4);
  float* gu     = (float*)alloc((size_t)T_ * 2 * (2 * F_) * 4);
  u16*   act    = (u16*)alloc((size_t)T_ * 2 * F_ * 2);
  float* eo     = (float*)alloc((size_t)T_ * 2 * D_ * 4);

  dim3 thr256(256);
  k_init<<<1, 32, 0, stream>>>(counts, cursor);

  // transpose+convert all weights: in[R][C] f32 -> out[C][R] bf16
  k_transpose_bf16<<<dim3(2048 / 32, D_ / 32, 1), thr256, 0, stream>>>(wq, wqT, D_, 2048);
  k_transpose_bf16<<<dim3(512 / 32, D_ / 32, 1), thr256, 0, stream>>>(wk, wkT, D_, 512);
  k_transpose_bf16<<<dim3(512 / 32, D_ / 32, 1), thr256, 0, stream>>>(wv, wvT, D_, 512);
  k_transpose_bf16<<<dim3(512 / 32, D_ / 32, 1), thr256, 0, stream>>>(wk_txt, wktT, D_, 512);
  k_transpose_bf16<<<dim3(512 / 32, D_ / 32, 1), thr256, 0, stream>>>(wv_txt, wvtT, D_, 512);
  k_transpose_bf16<<<dim3(D_ / 32, 2048 / 32, 1), thr256, 0, stream>>>(wo, woT, 2048, D_);
  k_transpose_bf16<<<dim3((2 * F_) / 32, D_ / 32, E_), thr256, 0, stream>>>(gup, gupT, D_, 2 * F_);
  k_transpose_bf16<<<dim3(D_ / 32, F_ / 32, E_), thr256, 0, stream>>>(dwn, dwnT, F_, D_);

  k_f32_to_bf16<<<dim3((unsigned)(((long long)B_ * ST_ * D_ + 255) / 256)), thr256, 0, stream>>>(
      enc, enc_bf, (long long)B_ * ST_ * D_);
  k_rmsnorm_bf16<<<dim3(T_), thr256, 0, stream>>>(hidden, norm1_w, x_bf, D_);

  auto gemm = [&](const u16* A, const u16* Bt, const float* res, float* C, int M, int N, int K) {
    k_gemm<<<dim3(N / 128, M / 128), thr256, 0, stream>>>(A, Bt, res, C, M, N, K);
  };
  gemm(x_bf,   wqT,  nullptr, qf,    T_,       H_ * HD_,  D_);
  gemm(x_bf,   wkT,  nullptr, kimgf, T_,       KV_ * HD_, D_);
  gemm(x_bf,   wvT,  nullptr, vimgf, T_,       KV_ * HD_, D_);
  gemm(enc_bf, wktT, nullptr, ktxtf, B_ * ST_, KV_ * HD_, D_);
  gemm(enc_bf, wvtT, nullptr, vtxtf, B_ * ST_, KV_ * HD_, D_);

  k_rms_rope<<<dim3(B_ * SI_ * H_), dim3(64), 0, stream>>>(
      qf, q_norm_w, img_cos, img_sin, q_bf, H_, SI_,
      (long long)SI_ * H_ * HD_, H_ * HD_, 0);
  k_rms_rope<<<dim3(B_ * SI_ * KV_), dim3(64), 0, stream>>>(
      kimgf, k_norm_w, img_cos, img_sin, k_bf, KV_, SI_,
      (long long)SKV_ * KV_ * HD_, KV_ * HD_, 0);
  k_rms_rope<<<dim3(B_ * ST_ * KV_), dim3(64), 0, stream>>>(
      ktxtf, akn_w, txt_cos, txt_sin, k_bf, KV_, ST_,
      (long long)SKV_ * KV_ * HD_, KV_ * HD_, SI_);
  k_copy_v<<<dim3(B_ * SI_), thr256, 0, stream>>>(vimgf, v_bf, SI_, (long long)SKV_ * KV_ * HD_, 0);
  k_copy_v<<<dim3(B_ * ST_), thr256, 0, stream>>>(vtxtf, v_bf, ST_, (long long)SKV_ * KV_ * HD_, SI_);

  k_attn<<<dim3(B_ * H_ * (SI_ / 64)), dim3(128), 0, stream>>>(q_bf, k_bf, v_bf, o_bf);

  gemm(o_bf, woT, hidden, out, T_, D_, H_ * HD_);   // h = hidden + o @ wo

  k_router<<<dim3(T_), thr256, 0, stream>>>(out, norm2_w, router_w, xt_bf, topi, topw, counts);
  k_offsets<<<1, 1, 0, stream>>>(counts, offs);
  k_scatter<<<dim3((T_ + 255) / 256), thr256, 0, stream>>>(topi, offs, cursor, perm);

  k_moe_gate_up<<<dim3((2 * F_) / 128, T_ / 128, E_), thr256, 0, stream>>>(
      xt_bf, gupT, counts, offs, perm, gu);
  k_silu<<<dim3((T_ * 2 * F_ / 2 + 255) / 256), thr256, 0, stream>>>(gu, act);
  k_moe_down<<<dim3(D_ / 128, T_ / 128, E_), thr256, 0, stream>>>(
      act, dwnT, counts, offs, perm, eo);
  k_final<<<dim3((T_ * D_ + 255) / 256), thr256, 0, stream>>>(out, eo, topw);
}
